// DockPointNet_83219286327861
// MI455X (gfx1250) — matrix operand order repaired and checked by reading, not compiled, not run on hardware
//
#include <hip/hip_runtime.h>
#include <hip/hip_bf16.h>
#include <math.h>

typedef _Float16 v16h __attribute__((ext_vector_type(16)));
typedef float    v8f  __attribute__((ext_vector_type(8)));

#define IN_CH 32
#define HIDC  47
#define OUTD  128
#define KPAD  64   // K padded 47 -> 64 (two K=32 WMMA steps)
#define N1PAD 48   // layer-1 N padded 47 -> 48 (three 16-wide tiles)

union FragU { v16h v; uint32_t u[8]; };

// 16-bit A-matrix 16x32 layout (wave32):
// lanes 0-15 : VGPR v holds K = hx*8 + 2v,2v+1 for v<4 ; K = 16 + hx*8 + ... for v>=4
__device__ __forceinline__ v16h load_a_frag(const _Float16* rowPtr, int hx, int c) {
  FragU A;
  const uint32_t* b = (const uint32_t*)rowPtr;  // 64 halves = 32 dwords per row
  const int kb = c * 32 + hx * 8;
  #pragma unroll
  for (int v = 0; v < 4; ++v) A.u[v]     = b[(kb + 2 * v) >> 1];
  #pragma unroll
  for (int v = 0; v < 4; ++v) A.u[4 + v] = b[(kb + 16 + 2 * v) >> 1];
  return A.v;
}

// 16-bit B-matrix 32x16 layout (wave32): VGPR v, lane half hx, col = lane&15:
// K pair = hx*8 + 2v, 2v+1 for v<4 ; K = 16 + hx*8 + ... for v>=4. Weights stored transposed (row = n, 64 contiguous k's).
__device__ __forceinline__ v16h load_b_frag(const _Float16* wRowPtr, int hx, int c) {
  FragU B;
  const uint32_t* b = (const uint32_t*)wRowPtr;
  const int kb = c * 32 + hx * 8;
  #pragma unroll
  for (int v = 0; v < 8; ++v) B.u[v] = b[(kb + (v >> 2) * 16 + 2 * (v & 3)) >> 1];
  return B.v;
}

__device__ __forceinline__ float angle3(float ax, float ay, float az,
                                        float bx, float by, float bz) {
  float cx = ay * bz - az * by;
  float cy = az * bx - ax * bz;
  float cz = ax * by - ay * bx;
  float cn = sqrtf(cx * cx + cy * cy + cz * cz);
  float d  = ax * bx + ay * by + az * bz;
  return atan2f(cn, d);
}

__global__ void dpnet_init_out(float* out, int n) {
  int i = blockIdx.x * 256 + threadIdx.x;
  if (i < n) out[i] = -INFINITY;
}

__global__ void dpnet_finalize_out(float* out, int n) {
  int i = blockIdx.x * 256 + threadIdx.x;
  if (i < n) {
    float v = out[i];
    out[i] = __builtin_isfinite(v) ? v : 0.0f;
  }
}

__global__ __launch_bounds__(128)
void dpnet_edge_kernel(const float* __restrict__ x,   const float* __restrict__ pos,
                       const float* __restrict__ nrm, const float* __restrict__ eattr,
                       const float* __restrict__ W1,  const float* __restrict__ b1,
                       const float* __restrict__ g1,  const float* __restrict__ be1,
                       const float* __restrict__ W2,  const float* __restrict__ b2,
                       const float* __restrict__ g2,  const float* __restrict__ be2,
                       const int* __restrict__ ei,    float* __restrict__ out,
                       int E) {
  __shared__ _Float16 sW1[N1PAD][KPAD];     // W1^T, f16, zero-padded
  __shared__ _Float16 sW2[OUTD][KPAD];      // W2^T, f16, zero-padded
  __shared__ float sB1[N1PAD], sG1[HIDC], sBe1[HIDC];
  __shared__ float sB2[OUTD],  sG2[OUTD],  sBe2[OUTD];
  __shared__ _Float16 sA[4][16][KPAD];      // per-wave A tile (msg, then normed h1)
  __shared__ float sH1[4][16][N1PAD];       // layer-1 relu output
  __shared__ float sH2[4][16][OUTD];        // layer-2 relu output
  __shared__ int   sDst[4][16];
  __shared__ float sMu[4][16], sRstd[4][16];

  const int tid  = threadIdx.x;
  const int wave = tid >> 5;
  const int lane = tid & 31;
  const int e    = lane & 15;   // edge row within tile / column within N-tile
  const int hx   = lane >> 4;   // half-wave index

  // ---- P0: stage weights/bias/affine into LDS (f16, padded) ----
  for (int i = tid; i < N1PAD * KPAD; i += 128) {
    int n = i >> 6, k = i & 63;
    sW1[n][k] = (_Float16)((n < HIDC && k < HIDC) ? W1[k * HIDC + n] : 0.0f);
  }
  for (int i = tid; i < OUTD * KPAD; i += 128) {
    int n = i >> 6, k = i & 63;
    sW2[n][k] = (_Float16)((k < HIDC) ? W2[k * OUTD + n] : 0.0f);
  }
  if (tid < N1PAD) sB1[tid] = (tid < HIDC) ? b1[tid] : 0.0f;
  if (tid < HIDC)  { sG1[tid] = g1[tid]; sBe1[tid] = be1[tid]; }
  if (tid < OUTD)  { sB2[tid] = b2[tid]; sG2[tid] = g2[tid]; sBe2[tid] = be2[tid]; }
  __syncthreads();

  const int  ge     = blockIdx.x * 64 + wave * 16 + e;
  const bool evalid = (ge < E);
  const int  src    = evalid ? ei[ge] : 0;

  // ---- P1: build msg tile [16 x 64] f16 ----
  {
    // x gather: each lane loads 16 of the 32 input channels for its edge
    const float* xr = x + (size_t)src * IN_CH + hx * 16;
    #pragma unroll
    for (int i = 0; i < 16; i += 4) {
      float4 v = *(const float4*)(xr + i);
      sA[wave][e][hx * 16 + i + 0] = (_Float16)v.x;
      sA[wave][e][hx * 16 + i + 1] = (_Float16)v.y;
      sA[wave][e][hx * 16 + i + 2] = (_Float16)v.z;
      sA[wave][e][hx * 16 + i + 3] = (_Float16)v.w;
    }
  }
  if (lane < 16) {
    int dst = evalid ? ei[E + ge] : -1;
    sDst[wave][e] = dst;
    float pix = 0.f, piy = 0.f, piz = 0.f, pjx = 0.f, pjy = 0.f, pjz = 0.f;
    float nix = 0.f, niy = 0.f, niz = 0.f, njx = 0.f, njy = 0.f, njz = 0.f;
    if (evalid) {
      pix = pos[dst * 3 + 0]; piy = pos[dst * 3 + 1]; piz = pos[dst * 3 + 2];
      pjx = pos[src * 3 + 0]; pjy = pos[src * 3 + 1]; pjz = pos[src * 3 + 2];
      nix = nrm[dst * 3 + 0]; niy = nrm[dst * 3 + 1]; niz = nrm[dst * 3 + 2];
      njx = nrm[src * 3 + 0]; njy = nrm[src * 3 + 1]; njz = nrm[src * 3 + 2];
    }
    float ux = pjx - pix, uy = pjy - piy, uz = pjz - piz;
    float dist = sqrtf(ux * ux + uy * uy + uz * uz);
    sA[wave][e][32] = (_Float16)dist;
    sA[wave][e][33] = (_Float16)angle3(nix, niy, niz, ux, uy, uz);
    sA[wave][e][34] = (_Float16)angle3(njx, njy, njz, ux, uy, uz);
    sA[wave][e][35] = (_Float16)angle3(nix, niy, niz, njx, njy, njz);
    #pragma unroll
    for (int i = 0; i < 11; ++i)
      sA[wave][e][36 + i] = (_Float16)(evalid ? eattr[(size_t)ge * 11 + i] : 0.0f);
    #pragma unroll
    for (int i = HIDC; i < KPAD; ++i) sA[wave][e][i] = (_Float16)0.0f;
  }
  __syncthreads();

  // ---- P2: layer 1 WMMA  [16x64] @ [64x48] -> [16x48] f32 ----
  const v8f vzero = {0.f, 0.f, 0.f, 0.f, 0.f, 0.f, 0.f, 0.f};
  v8f acc1[3];
  #pragma unroll
  for (int t = 0; t < 3; ++t) acc1[t] = vzero;
  #pragma unroll
  for (int c = 0; c < 2; ++c) {
    v16h a = load_a_frag(&sA[wave][e][0], hx, c);
    #pragma unroll
    for (int t = 0; t < 3; ++t) {
      v16h b = load_b_frag(&sW1[t * 16 + e][0], hx, c);
      acc1[t] = __builtin_amdgcn_wmma_f32_16x16x32_f16(
          false, a, false, b, (short)0, acc1[t], false, false);
    }
  }
  // D layout: VGPR r, lanes 0-15 -> M=r, lanes 16-31 -> M=r+8; N = lane&15 within tile
  #pragma unroll
  for (int t = 0; t < 3; ++t) {
    int n = t * 16 + e;
    float bias = sB1[n];
    #pragma unroll
    for (int r = 0; r < 8; ++r) {
      float v = acc1[t][r] + bias;
      sH1[wave][r + hx * 8][n] = v > 0.f ? v : 0.f;
    }
  }
  __syncthreads();

  // ---- P3: LayerNorm over 47 feats, repack as f16 A tile ----
  if (lane < 16) {
    float mu = 0.f;
    #pragma unroll
    for (int k = 0; k < HIDC; ++k) mu += sH1[wave][e][k];
    mu *= (1.0f / HIDC);
    float var = 0.f;
    #pragma unroll
    for (int k = 0; k < HIDC; ++k) { float d = sH1[wave][e][k] - mu; var += d * d; }
    var *= (1.0f / HIDC);
    float rstd = rsqrtf(var + 1e-5f);
    #pragma unroll
    for (int k = 0; k < HIDC; ++k)
      sA[wave][e][k] = (_Float16)((sH1[wave][e][k] - mu) * rstd * sG1[k] + sBe1[k]);
    #pragma unroll
    for (int k = HIDC; k < KPAD; ++k) sA[wave][e][k] = (_Float16)0.0f;
  }
  __syncthreads();

  // ---- P4: layer 2 WMMA  [16x64] @ [64x128] -> [16x128] f32 ----
  v8f acc2[8];
  #pragma unroll
  for (int t = 0; t < 8; ++t) acc2[t] = vzero;
  #pragma unroll
  for (int c = 0; c < 2; ++c) {
    v16h a = load_a_frag(&sA[wave][e][0], hx, c);
    #pragma unroll
    for (int t = 0; t < 8; ++t) {
      v16h b = load_b_frag(&sW2[t * 16 + e][0], hx, c);
      acc2[t] = __builtin_amdgcn_wmma_f32_16x16x32_f16(
          false, a, false, b, (short)0, acc2[t], false, false);
    }
  }
  #pragma unroll
  for (int t = 0; t < 8; ++t) {
    int n = t * 16 + e;
    float bias = sB2[n];
    #pragma unroll
    for (int r = 0; r < 8; ++r) {
      float v = acc2[t][r] + bias;
      sH2[wave][r + hx * 8][n] = v > 0.f ? v : 0.f;
    }
  }
  __syncthreads();

  // ---- P5: LayerNorm stats over 128 feats ----
  if (lane < 16) {
    float mu = 0.f;
    #pragma unroll 8
    for (int k = 0; k < OUTD; ++k) mu += sH2[wave][e][k];
    mu *= (1.0f / OUTD);
    float var = 0.f;
    #pragma unroll 8
    for (int k = 0; k < OUTD; ++k) { float d = sH2[wave][e][k] - mu; var += d * d; }
    var *= (1.0f / OUTD);
    sMu[wave][e]   = mu;
    sRstd[wave][e] = rsqrtf(var + 1e-5f);
  }
  __syncthreads();

  // ---- P6: affine + scatter-max (device-scope f32 atomic max, L2-resident) ----
  #pragma unroll 4
  for (int j = 0; j < 64; ++j) {
    int idx = lane + 32 * j;       // 16 edges x 128 feats = 2048 per wave
    int ee  = idx >> 7;
    int k   = idx & 127;
    int d   = sDst[wave][ee];
    if (d >= 0) {
      float v = (sH2[wave][ee][k] - sMu[wave][ee]) * sRstd[wave][ee] * sG2[k] + sBe2[k];
      __hip_atomic_fetch_max(out + (size_t)d * OUTD + k, v,
                             __ATOMIC_RELAXED, __HIP_MEMORY_SCOPE_AGENT);
    }
  }
}

extern "C" void kernel_launch(void* const* d_in, const int* in_sizes, int n_in,
                              void* d_out, int out_size, void* d_ws, size_t ws_size,
                              hipStream_t stream) {
  const float* x     = (const float*)d_in[0];
  const float* pos   = (const float*)d_in[1];
  const float* nrm   = (const float*)d_in[2];
  const float* eattr = (const float*)d_in[3];
  const float* W1    = (const float*)d_in[4];
  const float* b1    = (const float*)d_in[5];
  const float* g1    = (const float*)d_in[6];
  const float* be1   = (const float*)d_in[7];
  const float* W2    = (const float*)d_in[8];
  const float* b2    = (const float*)d_in[9];
  const float* g2    = (const float*)d_in[10];
  const float* be2   = (const float*)d_in[11];
  const int*   ei    = (const int*)d_in[12];
  float* out = (float*)d_out;

  const int E = in_sizes[12] / 2;

  dpnet_init_out<<<(out_size + 255) / 256, 256, 0, stream>>>(out, out_size);
  dpnet_edge_kernel<<<(E + 63) / 64, 128, 0, stream>>>(
      x, pos, nrm, eattr, W1, b1, g1, be1, W2, b2, g2, be2, ei, out, E);
  dpnet_finalize_out<<<(out_size + 255) / 256, 256, 0, stream>>>(out, out_size);
}
